// SinkhornScorer_49847390437965
// MI455X (gfx1250) — matrix operand order repaired and checked by reading, not compiled, not run on hardware
//
#include <hip/hip_runtime.h>
#include <hip/hip_bf16.h>
#include <math.h>

typedef __attribute__((ext_vector_type(16))) _Float16 v16h;
typedef __attribute__((ext_vector_type(8)))  _Float16 v8h;
typedef __attribute__((ext_vector_type(4)))  _Float16 v4h;
typedef __attribute__((ext_vector_type(8)))  float    v8f;

#define B_   64
#define NS_  512
#define H_   1024
#define NEG_LOG_N (-6.2383246250395075f)   // -log(512)
#define INV_REG   10.0f                    // 1/0.1

// ---------------------------------------------------------------------------
// Kernel 1: L2-normalize each 1024-float row and emit _Float16.
// One block (128 threads) per row. float4 loads (b128), v4h stores.
// ---------------------------------------------------------------------------
__global__ __launch_bounds__(128) void normalize_f16_kernel(
    const float* __restrict__ in, _Float16* __restrict__ out) {
  __shared__ float warpSums[4];
  const int row = blockIdx.x;
  const float* src = in + (size_t)row * H_;
  _Float16*    dst = out + (size_t)row * H_;
  const int t = threadIdx.x;

  float4 a = ((const float4*)src)[t];          // elems 4t .. 4t+3
  float4 b = ((const float4*)src)[t + 128];    // elems 512+4t ..
  float ss = a.x*a.x + a.y*a.y + a.z*a.z + a.w*a.w
           + b.x*b.x + b.y*b.y + b.z*b.z + b.w*b.w;
  #pragma unroll
  for (int off = 16; off >= 1; off >>= 1) ss += __shfl_xor(ss, off, 32);
  if ((t & 31) == 0) warpSums[t >> 5] = ss;
  __syncthreads();
  const float tot = warpSums[0] + warpSums[1] + warpSums[2] + warpSums[3];
  const float scale = 1.0f / fmaxf(sqrtf(tot), 1e-12f);

  v4h ha = { (_Float16)(a.x*scale), (_Float16)(a.y*scale),
             (_Float16)(a.z*scale), (_Float16)(a.w*scale) };
  v4h hb = { (_Float16)(b.x*scale), (_Float16)(b.y*scale),
             (_Float16)(b.z*scale), (_Float16)(b.w*scale) };
  ((v4h*)dst)[t]       = ha;
  ((v4h*)dst)[t + 128] = hb;
}

// ---------------------------------------------------------------------------
// Kernel 2: batched GEMM  S[b] = xn[b] (512x1024) * yn[b]^T (1024x512), fp32 out.
// Register-blocked: each wave owns a 64x64 macro-tile = 4x4 fragments of
// v_wmma_f32_16x16x32_f16.  Per K-chunk (32): 8 fragment loads -> 16 WMMAs
// (1 b128-pair per WMMA), 4x the data reuse of one-tile-per-wave.
// 64 batches * 8*8 macro-tiles = 4096 waves -> 512 blocks of 8 waves.
// ---------------------------------------------------------------------------
__global__ __launch_bounds__(256) void gemm_wmma_kernel(
    const _Float16* __restrict__ xh, const _Float16* __restrict__ yh,
    float* __restrict__ S) {
  const int tid  = threadIdx.x;
  const int lane = tid & 31;
  const int wave = tid >> 5;

  const int tileIdx = blockIdx.x * 8 + wave;
  const int b       = tileIdx >> 6;           // 64 macro-tiles per batch
  const int within  = tileIdx & 63;
  const int m0      = (within >> 3) << 6;     // 8 row macro-tiles of 64
  const int n0      = (within & 7) << 6;      // 8 col macro-tiles of 64

  const int half = lane >> 4;                 // 0: lanes 0-15, 1: lanes 16-31
  const int l16  = lane & 15;

  const size_t batchOff = (size_t)b * NS_ * H_;
  const int aShift = half * 16;               // A per-lane contiguous K chunk
  const int bShift = half * 16;               // B per-lane contiguous K chunk

  // A-matrix fragment rows: lane = M; B-matrix fragment cols: lane = N
  const _Float16* Ap[4];
  const _Float16* Bp[4];
  #pragma unroll
  for (int i = 0; i < 4; ++i) {
    Ap[i] = xh + batchOff + (size_t)(m0 + 16 * i + l16) * H_ + aShift;
    Bp[i] = yh + batchOff + (size_t)(n0 + 16 * i + l16) * H_ + bShift;
  }

  v8f c[4][4] = {};

  for (int kc = 0; kc < H_; kc += 32) {
    v16h av[4], bv[4];
    #pragma unroll
    for (int i = 0; i < 4; ++i) {
      union { v16h v; v8h h[2]; } a;
      a.h[0] = *(const v8h*)(Ap[i] + kc);
      a.h[1] = *(const v8h*)(Ap[i] + kc + 8);
      av[i]  = a.v;
      bv[i]  = *(const v16h*)(Bp[i] + kc);
    }
    if (kc + 32 < H_) {
      #pragma unroll
      for (int i = 0; i < 4; ++i) {
        __builtin_prefetch(Ap[i] + kc + 32, 0, 1);
        __builtin_prefetch(Bp[i] + kc + 32, 0, 1);
      }
    }
    #pragma unroll
    for (int im = 0; im < 4; ++im) {
      #pragma unroll
      for (int in = 0; in < 4; ++in) {
        // (neg_a, A, neg_b, B, c_mod, C, reuse_a, reuse_b)
        c[im][in] = __builtin_amdgcn_wmma_f32_16x16x32_f16(
            false, av[im], false, bv[in], (short)0, c[im][in], false, false);
      }
    }
  }

  // D layout per fragment: VGPR r -> row 16*im + r + 8*half, col 16*in + l16
  float* Sb = S + (size_t)b * NS_ * NS_;
  #pragma unroll
  for (int im = 0; im < 4; ++im) {
    #pragma unroll
    for (int in = 0; in < 4; ++in) {
      float* Srow = Sb + (size_t)(m0 + 16 * im + half * 8) * NS_
                       + (n0 + 16 * in + l16);
      #pragma unroll
      for (int r = 0; r < 8; ++r) Srow[(size_t)r * NS_] = c[im][in][r];
    }
  }
}

// ---------------------------------------------------------------------------
// Kernel 3: IPOT iterations + weighted sum, one block (1024 thr) per batch.
// Telescoped form: T_k = k*A + U_k(i) + V_k(j), state in LDS.
//   row pass:  U[i] = -log(n) - LSE_j( t*10*S(i,j) + V[j] + vprev[j] )
//   col pass:  newV[j] = -log(n) - LSE_i( t*10*S(i,j) + U[i] );
//              vprev = newV - V; V = newV
// final: out[b] = sum_ij S * exp(100*S + U(i) + V(j))
// ---------------------------------------------------------------------------
__global__ __launch_bounds__(1024) void sinkhorn_kernel(
    const float* __restrict__ S, float* __restrict__ out) {
  __shared__ float U[NS_], V[NS_], vprev[NS_];
  __shared__ float red[32];

  const int b    = blockIdx.x;
  const float* Sb = S + (size_t)b * NS_ * NS_;
  const int tid  = threadIdx.x;
  const int lane = tid & 31;
  const int wave = tid >> 5;

  if (tid < NS_) { U[tid] = 0.0f; V[tid] = 0.0f; vprev[tid] = NEG_LOG_N; }
  __syncthreads();

  for (int t = 1; t <= 10; ++t) {
    const float tA = INV_REG * (float)t;

    // ---- row pass: one wave per row, online LSE over 512 cols ----
    for (int row = wave; row < NS_; row += 32) {
      const float* Sr = Sb + (size_t)row * NS_;
      float m = -3.0e38f, s = 0.0f;
      for (int j = lane; j < NS_; j += 32) {
        const float v = tA * Sr[j] + V[j] + vprev[j];
        if (v > m) { s = s * __expf(m - v) + 1.0f; m = v; }
        else       { s += __expf(v - m); }
      }
      #pragma unroll
      for (int off = 16; off >= 1; off >>= 1) {
        const float mo = __shfl_xor(m, off, 32);
        const float so = __shfl_xor(s, off, 32);
        const float mn = fmaxf(m, mo);
        s = s * __expf(m - mn) + so * __expf(mo - mn);
        m = mn;
      }
      if (lane == 0) U[row] = NEG_LOG_N - (m + __logf(s));
    }
    __syncthreads();

    // ---- col pass: waves 0..15, lane = column, loop rows (coalesced) ----
    if (wave < 16) {
      const int j = wave * 32 + lane;
      float m = -3.0e38f, s = 0.0f;
      for (int i = 0; i < NS_; ++i) {
        const float v = tA * Sb[(size_t)i * NS_ + j] + U[i];
        if (v > m) { s = s * __expf(m - v) + 1.0f; m = v; }
        else       { s += __expf(v - m); }
      }
      const float newV = NEG_LOG_N - (m + __logf(s));
      vprev[j] = newV - V[j];
      V[j]     = newV;
    }
    __syncthreads();
  }

  // ---- final: sum_ij S * exp(100*S + U(i) + V(j)) ----
  float acc = 0.0f;
  for (int idx = tid; idx < NS_ * NS_; idx += 1024) {
    const int i = idx >> 9, j = idx & (NS_ - 1);
    const float sv = Sb[idx];
    acc += sv * __expf(100.0f * sv + U[i] + V[j]);
  }
  #pragma unroll
  for (int off = 16; off >= 1; off >>= 1) acc += __shfl_xor(acc, off, 32);
  if (lane == 0) red[wave] = acc;
  __syncthreads();
  if (wave == 0) {
    float v = red[lane];
    #pragma unroll
    for (int off = 16; off >= 1; off >>= 1) v += __shfl_xor(v, off, 32);
    if (lane == 0) out[b] = v;
  }
}

// ---------------------------------------------------------------------------
extern "C" void kernel_launch(void* const* d_in, const int* in_sizes, int n_in,
                              void* d_out, int out_size, void* d_ws, size_t ws_size,
                              hipStream_t stream) {
  const float* x = (const float*)d_in[0];
  const float* y = (const float*)d_in[1];

  char* ws = (char*)d_ws;
  const size_t f16Bytes = (size_t)B_ * NS_ * H_ * sizeof(_Float16);  // 64 MB
  _Float16* xh = (_Float16*)ws;
  _Float16* yh = (_Float16*)(ws + f16Bytes);
  float*    S  = (float*)(ws + 2 * f16Bytes);                         // 64 MB fp32

  // 1) normalize + fp32 -> f16
  normalize_f16_kernel<<<B_ * NS_, 128, 0, stream>>>(x, xh);
  normalize_f16_kernel<<<B_ * NS_, 128, 0, stream>>>(y, yh);

  // 2) batched cosine GEMM via WMMA: 4096 macro-tiles / 8 waves per block
  gemm_wmma_kernel<<<(B_ * 8 * 8) / 8, 256, 0, stream>>>(xh, yh, S);

  // 3) IPOT iterations + weighted sum, one workgroup per batch
  sinkhorn_kernel<<<B_, 1024, 0, stream>>>(S, (float*)d_out);
}
